// Comba_19430432047451
// MI455X (gfx1250) — compile-verified, matrix operand order checked by reading, not run on hardware
//
#include <hip/hip_runtime.h>
#include <hip/hip_bf16.h>

// ---------------------------------------------------------------------------
// Comba (IPLR linear attention) forward for MI455X / gfx1250.
//   * All matmuls via v_wmma_f32_16x16x32_bf16; every WMMA operand stored as
//     bf16, pair-packed TRANSPOSED where the contraction dim is non-contiguous
//     so both A and B fragments load as global_load_b128 pairs.
//   * Recurrence in exact chunked form (C=64); per-chunk matrices precomputed
//     in parallel ((I+A)^-1 folded into W2/Ubase, gamma-scaled Qg/KpT); the
//     sequential kernel keeps the 256x64 fp32 state + bf16 shadow in LDS
//     (320KB/WGP) and double-buffers next-chunk W2 via the Tensor Data Mover.
// ---------------------------------------------------------------------------

typedef __attribute__((ext_vector_type(16))) __bf16           v16bf;
typedef __attribute__((ext_vector_type(8)))  float            v8f;
typedef __attribute__((ext_vector_type(8)))  unsigned         v8u;
typedef __attribute__((ext_vector_type(8)))  unsigned short   v8us;

#define DEV static __device__ __forceinline__

static constexpr int Bc = 2, Tc = 2048, DM = 2048, Hh = 6, HK = 256, HV = 512;
static constexpr int KD = Hh * HK;           // 1536
static constexpr int VD = Hh * HV;           // 3072
static constexpr int BT = Bc * Tc;           // 4096
static constexpr int CH = 64, NC = Tc / CH;  // chunk 64, 32 chunks

DEV unsigned short f2bf_u(float f) {
  unsigned u = __builtin_bit_cast(unsigned, f);
  return (unsigned short)((u + 0x7FFFu + ((u >> 16) & 1u)) >> 16);
}
DEV float bfu_to_f(unsigned short h) {
  return __builtin_bit_cast(float, (unsigned)h << 16);
}
DEV __bf16 bf_from_f(float f) { return __builtin_bit_cast(__bf16, f2bf_u(f)); }
DEV float sigmoidf_(float x) { return 1.0f / (1.0f + expf(-x)); }

DEV v8f v8zero() {
  v8f z;
#pragma unroll
  for (int j = 0; j < 8; ++j) z[j] = 0.0f;
  return z;
}
DEV v8f wmma_bf16(v16bf a, v16bf b, v8f c) {
  return __builtin_amdgcn_wmma_f32_16x16x32_bf16(false, a, false, b, (short)0, c,
                                                 false, false);
}

// bf16-pair word index inside a 16x32 fragment for vgpr j, lane-half hi
DEV int frag_w(int j, int hi) { return ((j >> 2) << 3) + (hi << 2) + (j & 3); }
DEV int frag_k(int j, int hi) { return ((j >> 2) << 4) + (hi << 3) + ((j & 3) << 1); }

// A fragment from plain bf16 row-major: element (row0+r, kb+kk)
DEV v16bf load_a_bf(const unsigned short* src, int lda, int row0, int kb, int lane) {
  int r = lane & 15, hi = lane >> 4;
  const unsigned* rowp = (const unsigned*)(src + (long)(row0 + r) * lda + kb);
  v8u u;
#pragma unroll
  for (int j = 0; j < 8; ++j) u[j] = rowp[frag_w(j, hi)];
  return __builtin_bit_cast(v16bf, u);
}
// B fragment where memory holds the operand TRANSPOSED (bf16):
// element (kb+kk, col0+c) = src[(col0+c)*ld + kb+kk]   (pairs contiguous)
DEV v16bf load_bt_bf(const unsigned short* src, int ld, int kb, int col0, int lane) {
  int c = lane & 15, hi = lane >> 4;
  const unsigned* colp = (const unsigned*)(src + (long)(col0 + c) * ld + kb);
  v8u u;
#pragma unroll
  for (int j = 0; j < 8; ++j) u[j] = colp[frag_w(j, hi)];
  return __builtin_bit_cast(v16bf, u);
}
// B fragment from TRANSPOSED pair-packed words: Pt[n*kwords + k/2] = (k,k+1)
DEV v16bf load_bT_pk(const unsigned* Pt, long kwords, int kb, int col0, int lane) {
  int c = lane & 15, hi = lane >> 4;
  const unsigned* colp = Pt + (long)(col0 + c) * kwords + (kb >> 1);
  v8u u;
#pragma unroll
  for (int j = 0; j < 8; ++j) u[j] = colp[frag_w(j, hi)];
  return __builtin_bit_cast(v16bf, u);
}
// A fragment from fp32 with per-contraction-index scale (K4 Tinv only)
DEV v16bf load_a_f32_cs(const float* src, int lda, int row0, int kb,
                        const float* cs, int lane) {
  int r = lane & 15, hi = lane >> 4;
  const float* rowp = src + (row0 + r) * lda + kb;
  v16bf a;
#pragma unroll
  for (int j = 0; j < 8; ++j) {
    int k2 = frag_k(j, hi);
    a[2 * j]     = bf_from_f(rowp[k2] * cs[kb + k2]);
    a[2 * j + 1] = bf_from_f(rowp[k2 + 1] * cs[kb + k2 + 1]);
  }
  return a;
}
// fp32 C/D fragment: element (row0 + 8*hi + j, col0 + (lane&15))
DEV v8f load_cd(const float* src, int ldd, int row0, int col0, int lane) {
  int n = lane & 15, hi = lane >> 4;
  v8f c;
#pragma unroll
  for (int j = 0; j < 8; ++j) c[j] = src[(long)(row0 + hi * 8 + j) * ldd + col0 + n];
  return c;
}
DEV void store_cd(float* dst, int ldd, int row0, int col0, v8f c, int lane) {
  int n = lane & 15, hi = lane >> 4;
#pragma unroll
  for (int j = 0; j < 8; ++j) dst[(long)(row0 + hi * 8 + j) * ldd + col0 + n] = c[j];
}
// store C/D as bf16 into a COLUMN-MAJOR buffer: 8 consecutive rows -> 16B store
DEV void store_cd_bf_cm(unsigned short* dst, int ld, int row0, int col0, v8f c,
                        int lane) {
  int n = lane & 15, hi = lane >> 4;
  v8us pk;
#pragma unroll
  for (int j = 0; j < 8; ++j) pk[j] = f2bf_u(c[j]);
  *(v8us*)(dst + (long)(col0 + n) * ld + row0 + hi * 8) = pk;
}

// ------------------------- Tensor Data Mover ------------------------------
#if defined(__has_builtin)
#if __has_builtin(__builtin_amdgcn_tensor_load_to_lds)
#define HAVE_TDM 1
#endif
#endif
#ifndef HAVE_TDM
#define HAVE_TDM 0
#endif

#if HAVE_TDM
typedef __attribute__((ext_vector_type(4))) unsigned tdm_u32x4;
typedef __attribute__((ext_vector_type(8))) int      tdm_i32x8;
typedef __attribute__((ext_vector_type(4))) int      tdm_i32x4;

// 2-D contiguous bf16 tile: width elems per row, height rows.
DEV void tdm_load_2d_bf16(const void* gaddr, unsigned lds_byte_off,
                          unsigned width, unsigned height) {
  unsigned long long ga = (unsigned long long)gaddr;
  tdm_u32x4 g0;
  g0[0] = 1u;                                             // count=1 (valid D#)
  g0[1] = lds_byte_off;                                   // lds_addr
  g0[2] = (unsigned)(ga & 0xFFFFFFFFu);                   // global_addr lo
  g0[3] = (unsigned)((ga >> 32) & 0x01FFFFFFu) | (2u << 30);  // addr hi | type=2
  tdm_i32x8 g1;
  g1[0] = (int)(1u << 16);                                // data_size=1 (2B)
  g1[1] = (int)((width & 0xFFFFu) << 16);                 // tensor_dim0 lo
  g1[2] = (int)(((width >> 16) & 0xFFFFu) | ((height & 0xFFFFu) << 16));
  g1[3] = (int)((((height >> 16) & 0xFFFFu)) | ((width & 0xFFFFu) << 16)); // tile_dim0
  g1[4] = (int)(height & 0xFFFFu);                        // tile_dim1
  g1[5] = (int)width;                                     // dim0_stride lo
  g1[6] = 0;
  g1[7] = 0;
  tdm_i32x4 z4 = {0, 0, 0, 0};
#if __clang_major__ >= 23
  tdm_i32x8 z8 = {0, 0, 0, 0, 0, 0, 0, 0};
  __builtin_amdgcn_tensor_load_to_lds(g0, g1, z4, z4, z8, 0);
#else
  __builtin_amdgcn_tensor_load_to_lds(g0, g1, z4, z4, 0);
#endif
}
DEV unsigned lds_off_of(const void* p) {
  return (unsigned)(unsigned long long)p;  // low 32 bits = LDS offset
}
#endif

// ---------------------------------------------------------------------------
// Elementwise pack/convert kernels (one-time conversions)
// ---------------------------------------------------------------------------
__global__ void k_f2bf(const float* __restrict__ x, unsigned short* __restrict__ y,
                       long n) {
  long i = (long)blockIdx.x * blockDim.x + threadIdx.x;
  if (i < n) y[i] = f2bf_u(x[i]);
}
// fp32 [K x N] -> TRANSPOSED packed u32 [N x K/2]
__global__ void k_packwT(const float* __restrict__ W, unsigned* __restrict__ P,
                         int K, int N) {
  long i = (long)blockIdx.x * blockDim.x + threadIdx.x;
  long total = (long)N * (K / 2);
  if (i >= total) return;
  int kw = (int)(i % (K / 2));
  long n = i / (K / 2);
  unsigned lo = f2bf_u(W[(long)(2 * kw) * N + n]);
  unsigned hi = f2bf_u(W[(long)(2 * kw + 1) * N + n]);
  P[i] = lo | (hi << 16);
}
// bf16 [R x C] -> TRANSPOSED packed u32 [C x R/2]
__global__ void k_packrowsT(const unsigned short* __restrict__ S,
                            unsigned* __restrict__ P, int R, int C) {
  long i = (long)blockIdx.x * blockDim.x + threadIdx.x;
  long total = (long)C * (R / 2);
  if (i >= total) return;
  int rw = (int)(i % (R / 2));
  long c = i / (R / 2);
  P[i] = (unsigned)S[(long)(2 * rw) * C + c] |
         ((unsigned)S[(long)(2 * rw + 1) * C + c] << 16);
}

// ---------------------------------------------------------------------------
// K1: bf16-WMMA GEMM, compile-time shapes so addressing folds to immediates.
// A: plain bf16 [MxK]; Bt: transposed pair-packed u32 [N x K/2]; C fp32 [MxN].
// ---------------------------------------------------------------------------
template <int N, int K>
__global__ __launch_bounds__(256) void k_gemm_bf(
    const unsigned short* __restrict__ A, const unsigned* __restrict__ Bt,
    float* __restrict__ C) {
  int lane = threadIdx.x & 31, w = threadIdx.x >> 5;
  int wr = w >> 2, wc = w & 3;
  int row0 = blockIdx.y * 64 + wr * 32;
  int col0 = blockIdx.x * 128 + wc * 32;
  v8f acc00 = v8zero(), acc01 = v8zero(), acc10 = v8zero(), acc11 = v8zero();
  for (int kb = 0; kb < K; kb += 32) {
    v16bf a0 = load_a_bf(A, K, row0, kb, lane);
    v16bf a1 = load_a_bf(A, K, row0 + 16, kb, lane);
    v16bf b0 = load_bT_pk(Bt, K / 2, kb, col0, lane);
    v16bf b1 = load_bT_pk(Bt, K / 2, kb, col0 + 16, lane);
    acc00 = wmma_bf16(a0, b0, acc00);
    acc01 = wmma_bf16(a0, b1, acc01);
    acc10 = wmma_bf16(a1, b0, acc10);
    acc11 = wmma_bf16(a1, b1, acc11);
  }
  store_cd(C, N, row0, col0, acc00, lane);
  store_cd(C, N, row0, col0 + 16, acc01, lane);
  store_cd(C, N, row0 + 16, col0, acc10, lane);
  store_cd(C, N, row0 + 16, col0 + 16, acc11, lane);
}

// Tiny-N GEMM (Wa/Wb, N=6)
__global__ void k_gemm_small(const float* __restrict__ A, const float* __restrict__ B,
                             float* __restrict__ C, int M, int N, int K) {
  int i = blockIdx.x * blockDim.x + threadIdx.x;
  if (i >= M * N) return;
  int m = i / N, n = i % N;
  float s = 0.0f;
  for (int k = 0; k < K; ++k) s += A[m * K + k] * B[k * N + n];
  C[i] = s;
}

// K2: causal depthwise conv (K=4) + SiLU
__global__ void k_conv_silu(const float* __restrict__ x, const float* __restrict__ w,
                            float* __restrict__ y, int Cn) {
  long i = (long)blockIdx.x * blockDim.x + threadIdx.x;
  long total = (long)BT * Cn;
  if (i >= total) return;
  int c = (int)(i % Cn);
  long bt = i / Cn;
  int t = (int)(bt % Tc);
  float s = 0.0f;
#pragma unroll
  for (int j = 0; j < 4; ++j) {
    int ts = t - 3 + j;
    if (ts >= 0) s += x[i + (long)(ts - t) * Cn] * w[c * 4 + j];
  }
  y[i] = s * sigmoidf_(s);
}

// K3: per-(b,t,h) features -> bf16 q,k,p + fp32 g, beta
__global__ __launch_bounds__(256) void k_features(
    const float* __restrict__ qc, const float* __restrict__ kc,
    const float* __restrict__ decay, const float* __restrict__ Dp,
    const float* __restrict__ A_log, const float* __restrict__ dtb,
    const float* __restrict__ aw, const float* __restrict__ bw,
    unsigned short* __restrict__ qn, unsigned short* __restrict__ kn,
    unsigned short* __restrict__ pn, float* __restrict__ gv,
    float* __restrict__ bev) {
  __shared__ float red[256];
  int bth = blockIdx.x, h = bth % Hh, bt = bth / Hh;
  int tid = threadIdx.x;
  long base = (long)bt * KD + h * HK + tid;
  float kq = qc[base], kk = kc[base];
  float sd = sigmoidf_(decay[h]);
  float pv = kk * sd;
  float qv = kq - Dp[h] * pv;

  red[tid] = qv * qv;
  __syncthreads();
  for (int s = 128; s > 0; s >>= 1) { if (tid < s) red[tid] += red[tid + s]; __syncthreads(); }
  float sq = red[0];
  __syncthreads();
  red[tid] = kk * kk;
  __syncthreads();
  for (int s = 128; s > 0; s >>= 1) { if (tid < s) red[tid] += red[tid + s]; __syncthreads(); }
  float sk = red[0];

  qn[base] = f2bf_u(qv * rsqrtf(sq + 1e-6f) * 0.0625f);  // * head_k^-0.5
  kn[base] = f2bf_u(kk * rsqrtf(sk + 1e-6f));
  pn[base] = f2bf_u(pv);
  if (tid == 0) {
    float av = aw[(long)bt * Hh + h] + dtb[h];
    float sp = (av > 20.0f) ? av : log1pf(expf(av));
    gv[(long)bt * Hh + h] = -expf(A_log[h]) * sp;
    bev[(long)bt * Hh + h] = 2.0f * sigmoidf_(bw[(long)bt * Hh + h]);
  }
}

// ---------------------------------------------------------------------------
// K4: per (b,h,chunk) parallel precompute (all outputs WMMA-ready bf16):
//   Mb = tril(e^{cg_t-cg_s} q.k), Tinv=(I+A)^-1, Ubase=Tinv diag(b) Vc (fp32),
//   W2b = -(Tinv diag(b*gamma)) Pc, Qg = diag(gamma) Q, KpT[m][s]=kps_s k_s[m].
// ---------------------------------------------------------------------------
__global__ __launch_bounds__(256) void k_chunkprep(
    const unsigned short* __restrict__ qnb, const unsigned short* __restrict__ knb,
    const unsigned short* __restrict__ pnb, const unsigned* __restrict__ ppkT,
    const unsigned* __restrict__ vpkT, const float* __restrict__ gv,
    const float* __restrict__ bev, float* __restrict__ gCbuf,
    unsigned short* __restrict__ Mb, unsigned short* __restrict__ W2b,
    unsigned short* __restrict__ Qg, unsigned short* __restrict__ KpT,
    float* __restrict__ Ubase) {
  extern __shared__ float sm[];
  float* cgs  = sm;             // 64 cumulative g
  float* bsh  = sm + 64;        // 64 beta
  float* csA  = sm + 128;       // 64 beta_s
  float* csB  = sm + 192;       // 64 beta_s*gamma_s
  float* gamL = sm + 256;       // 64 exp(cg)
  float* kpsL = sm + 320;       // 64 exp(cg63-cg)
  float* Amat = sm + 384;       // 64*64
  float* Tinv = sm + 384 + CH * CH;  // 64*64

  int blk = blockIdx.x;
  int c = blk % NC, bh = blk / NC, h = bh % Hh, b = bh / Hh;
  int t0 = c * CH;
  int tid = threadIdx.x, lane = tid & 31, w = tid >> 5;

  const unsigned short* qb  = qnb + (long)(b * Tc + t0) * KD + h * HK;
  const unsigned short* kb_ = knb + (long)(b * Tc + t0) * KD + h * HK;
  const unsigned short* pb  = pnb + (long)(b * Tc + t0) * KD + h * HK;
  // transposed time-pair-packed operands: base at (channel row, time word)
  const unsigned* pbt = ppkT + (long)(h * HK) * (BT / 2) + ((b * Tc + t0) >> 1);
  const unsigned* vbt = vpkT + (long)(h * HV) * (BT / 2) + ((b * Tc + t0) >> 1);

  if (tid < CH) {
    cgs[tid] = gv[(long)(b * Tc + t0 + tid) * Hh + h];
    bsh[tid] = bev[(long)(b * Tc + t0 + tid) * Hh + h];
  }
  __syncthreads();
  if (tid == 0) {
    float s = 0.0f;
    for (int t = 0; t < CH; ++t) { s += cgs[t]; cgs[t] = s; }
  }
  __syncthreads();
  if (tid < CH) {
    float g = expf(cgs[tid]);
    csA[tid]  = bsh[tid];
    csB[tid]  = bsh[tid] * g;
    gamL[tid] = g;
    kpsL[tid] = expf(cgs[CH - 1] - cgs[tid]);
  }
  if (tid == 0) gCbuf[(long)bh * NC + c] = expf(cgs[CH - 1]);
  __syncthreads();

  // A (fp32, LDS) and M (bf16, global) via WMMA: [64x256]@[256x64]
  unsigned short* mptr = Mb + ((long)bh * NC + c) * CH * CH;
  for (int ti = w; ti < 16; ti += 8) {
    int tr = ti >> 2, sc = ti & 3;
    v8f accA = v8zero(), accM = v8zero();
#pragma unroll
    for (int kb2 = 0; kb2 < HK; kb2 += 32) {
      v16bf bk = load_bt_bf(kb_, KD, kb2, sc * 16, lane);  // K^T
      v16bf ap = load_a_bf(pb, KD, tr * 16, kb2, lane);
      v16bf aq = load_a_bf(qb, KD, tr * 16, kb2, lane);
      accA = wmma_bf16(ap, bk, accA);
      accM = wmma_bf16(aq, bk, accM);
    }
    int n = lane & 15, hi = lane >> 4;
#pragma unroll
    for (int j = 0; j < 8; ++j) {
      int t = tr * 16 + hi * 8 + j, s = sc * 16 + n;
      float e = expf(cgs[t] - cgs[s]);
      Amat[t * CH + s] = (s < t) ? bsh[t] * e * accA[j] : 0.0f;
      mptr[t * CH + s] = f2bf_u((s <= t) ? e * accM[j] : 0.0f);
    }
  }
  __syncthreads();

  // Tinv = (I + A)^-1 forward substitution
  for (int i = tid; i < CH * CH; i += 256) Tinv[i] = ((i / CH) == (i % CH)) ? 1.0f : 0.0f;
  __syncthreads();
  for (int t = 1; t < CH; ++t) {
    if (tid < CH) {
      float s = 0.0f;
      for (int u = 0; u < t; ++u) s += Amat[t * CH + u] * Tinv[u * CH + tid];
      Tinv[t * CH + tid] = ((tid == t) ? 1.0f : 0.0f) - s;
    }
    __syncthreads();
  }

  // Ubase = (Tinv diag(beta)) @ Vc -> fp32 [64 x 512]
  float* ub = Ubase + ((long)bh * NC + c) * CH * HV;
  for (int ti = w; ti < 128; ti += 8) {
    int tr = ti >> 5, vc = ti & 31;
    v8f acc = v8zero();
#pragma unroll
    for (int kb2 = 0; kb2 < CH; kb2 += 32) {
      v16bf a  = load_a_f32_cs(Tinv, CH, tr * 16, kb2, csA, lane);
      v16bf bf = load_bT_pk(vbt, BT / 2, kb2, vc * 16, lane);
      acc = wmma_bf16(a, bf, acc);
    }
    store_cd(ub, HV, tr * 16, vc * 16, acc, lane);
  }
  // W2b = -(Tinv diag(beta*gamma)) @ Pc -> bf16 [64 x 256]
  unsigned short* w2 = W2b + ((long)bh * NC + c) * CH * HK;
  for (int ti = w; ti < 64; ti += 8) {
    int tr = ti >> 4, kc2 = ti & 15;
    v8f acc = v8zero();
#pragma unroll
    for (int kb2 = 0; kb2 < CH; kb2 += 32) {
      v16bf a  = load_a_f32_cs(Tinv, CH, tr * 16, kb2, csB, lane);
      v16bf bf = load_bT_pk(pbt, BT / 2, kb2, kc2 * 16, lane);
      acc = wmma_bf16(a, bf, acc);
    }
    int n = lane & 15, hi = lane >> 4;
#pragma unroll
    for (int j = 0; j < 8; ++j)
      w2[(tr * 16 + hi * 8 + j) * HK + kc2 * 16 + n] = f2bf_u(-acc[j]);
  }

  // Qg[t][k] = gamma_t * q_t[k];  KpT[m][s] = kps_s * k_s[m]   (bf16)
  unsigned short* qg  = Qg + ((long)bh * NC + c) * CH * HK;
  unsigned short* kpt = KpT + ((long)bh * NC + c) * HK * CH;
  for (int i = tid; i < CH * HK; i += 256) {
    int t = i >> 8, kk = i & (HK - 1);
    qg[i] = f2bf_u(bfu_to_f(qb[t * KD + kk]) * gamL[t]);
  }
  for (int i = tid; i < HK * CH; i += 256) {
    int m = i >> 6, s = i & (CH - 1);
    kpt[i] = f2bf_u(bfu_to_f(kb_[s * KD + m]) * kpsL[s]);
  }
}

// ---------------------------------------------------------------------------
// K5: sequential-over-chunks recurrence. One block per (b,h,vblk).
// LDS: S fp32 64KB + Sb bf16 shadow 32KB + Ub bf16 8KB + W2 TDM stage 64KB.
// ---------------------------------------------------------------------------
__global__ __launch_bounds__(256) void k_recur(
    const unsigned short* __restrict__ Mb, const unsigned short* __restrict__ W2b,
    const unsigned short* __restrict__ Qg, const unsigned short* __restrict__ KpT,
    const float* __restrict__ Ubase, const float* __restrict__ gCbuf,
    float* __restrict__ obuf) {
  extern __shared__ char smc[];
  float* S = (float*)smc;                                   // [256][64] fp32
  unsigned short* Sb = (unsigned short*)(smc + 65536);      // col-major [64][256]
  unsigned short* Ub = (unsigned short*)(smc + 98304);      // col-major [64][64]
  unsigned short* stg = (unsigned short*)(smc + 106496);    // 2 x [64][256] W2 stage

  int blk = blockIdx.x;
  int vb = blk & 7, bh = blk >> 3, h = bh % Hh, b = bh / Hh;
  int v0 = vb * CH;
  int tid = threadIdx.x, lane = tid & 31, w = tid >> 5;

  for (int i = tid; i < HK * CH; i += 256) S[i] = 0.0f;
  for (int i = tid; i < HK * CH / 2; i += 256) ((unsigned*)Sb)[i] = 0u;
  __syncthreads();

#if HAVE_TDM
  if (w == 0)
    tdm_load_2d_bf16(W2b + (long)bh * NC * CH * HK, lds_off_of(stg), HK, CH);
#endif

  for (int c = 0; c < NC; ++c) {
    int t0 = c * CH;
    const unsigned short* mb  = Mb + ((long)bh * NC + c) * CH * CH;
    const unsigned short* qg  = Qg + ((long)bh * NC + c) * CH * HK;
    const unsigned short* kpt = KpT + ((long)bh * NC + c) * HK * CH;
    const float* ub = Ubase + ((long)bh * NC + c) * CH * HV + v0;
    float* ob = obuf + (long)(b * Tc + t0) * VD + h * HV + v0;
    float gC = gCbuf[(long)bh * NC + c];

#if HAVE_TDM
    if (w == 0) __builtin_amdgcn_s_wait_tensorcnt((short)0);
    __syncthreads();
    if (w == 0 && c + 1 < NC)
      tdm_load_2d_bf16(W2b + ((long)bh * NC + c + 1) * CH * HK,
                       lds_off_of(stg + ((c + 1) & 1) * CH * HK), HK, CH);
    const unsigned short* w2a = stg + (c & 1) * CH * HK;
#else
    const unsigned short* w2a = W2b + ((long)bh * NC + c) * CH * HK;
#endif
    if (tid == 0 && c + 1 < NC) {  // keep next chunk operands warm in L2
      __builtin_prefetch(Mb + ((long)bh * NC + c + 1) * CH * CH, 0, 1);
      __builtin_prefetch(Qg + ((long)bh * NC + c + 1) * CH * HK, 0, 1);
      __builtin_prefetch(KpT + ((long)bh * NC + c + 1) * HK * CH, 0, 1);
      __builtin_prefetch(Ubase + ((long)bh * NC + c + 1) * CH * HV + v0, 0, 1);
    }

    // Step A: U = Ubase + W2 @ S0   (store bf16 col-major into Ub)
    for (int ti = w; ti < 16; ti += 8) {
      int tr = ti >> 2, nc = ti & 3;
      v8f acc = load_cd(ub, HV, tr * 16, nc * 16, lane);
#pragma unroll
      for (int kb2 = 0; kb2 < HK; kb2 += 32) {
        v16bf a  = load_a_bf(w2a, HK, tr * 16, kb2, lane);
        v16bf bf = load_bt_bf(Sb, HK, kb2, nc * 16, lane);
        acc = wmma_bf16(a, bf, acc);
      }
      store_cd_bf_cm(Ub, CH, tr * 16, nc * 16, acc, lane);
    }
    __syncthreads();

    // Step B: O = M @ U + Qg @ S0
    for (int ti = w; ti < 16; ti += 8) {
      int tr = ti >> 2, nc = ti & 3;
      v8f acc = v8zero();
#pragma unroll
      for (int kb2 = 0; kb2 < CH; kb2 += 32) {
        v16bf a  = load_a_bf(mb, CH, tr * 16, kb2, lane);
        v16bf bf = load_bt_bf(Ub, CH, kb2, nc * 16, lane);
        acc = wmma_bf16(a, bf, acc);
      }
#pragma unroll
      for (int kb2 = 0; kb2 < HK; kb2 += 32) {
        v16bf a  = load_a_bf(qg, HK, tr * 16, kb2, lane);
        v16bf bf = load_bt_bf(Sb, HK, kb2, nc * 16, lane);
        acc = wmma_bf16(a, bf, acc);
      }
      store_cd(ob, VD, tr * 16, nc * 16, acc, lane);
    }
    __syncthreads();

    // Step C: S = gammaC*S + Kp^T @ U  (update fp32 S and bf16 shadow Sb)
    for (int ti = w; ti < 64; ti += 8) {
      int mr = ti >> 2, nc = ti & 3;
      v8f acc = load_cd(S, CH, mr * 16, nc * 16, lane);
#pragma unroll
      for (int j = 0; j < 8; ++j) acc[j] *= gC;
#pragma unroll
      for (int s0 = 0; s0 < CH; s0 += 32) {
        v16bf a  = load_a_bf(kpt, CH, mr * 16, s0, lane);
        v16bf bf = load_bt_bf(Ub, CH, s0, nc * 16, lane);
        acc = wmma_bf16(a, bf, acc);
      }
      store_cd(S, CH, mr * 16, nc * 16, acc, lane);
      store_cd_bf_cm(Sb, HK, mr * 16, nc * 16, acc, lane);
    }
    __syncthreads();
  }
}

// K6: FusedRMSNormGated(sigmoid) -> bf16 output (A-operand of final GEMM)
__global__ __launch_bounds__(256) void k_gatednorm(
    const float* __restrict__ obuf, const float* __restrict__ gate,
    const float* __restrict__ onw, unsigned short* __restrict__ out2b) {
  __shared__ float red[256];
  int bth = blockIdx.x, h = bth % Hh, bt = bth / Hh;
  int tid = threadIdx.x;
  long base = (long)bt * VD + h * HV;
  float x0 = obuf[base + tid], x1 = obuf[base + tid + 256];
  red[tid] = x0 * x0 + x1 * x1;
  __syncthreads();
  for (int s = 128; s > 0; s >>= 1) { if (tid < s) red[tid] += red[tid + s]; __syncthreads(); }
  float sc = rsqrtf(red[0] / (float)HV + 1e-5f);
  out2b[base + tid]       = f2bf_u(x0 * sc * onw[tid]       * sigmoidf_(gate[base + tid]));
  out2b[base + tid + 256] = f2bf_u(x1 * sc * onw[tid + 256] * sigmoidf_(gate[base + tid + 256]));
}

// ---------------------------------------------------------------------------
extern "C" void kernel_launch(void* const* d_in, const int* in_sizes, int n_in,
                              void* d_out, int out_size, void* d_ws, size_t ws_size,
                              hipStream_t stream) {
  (void)in_sizes; (void)n_in; (void)out_size; (void)ws_size;
  const float* hs    = (const float*)d_in[0];
  const float* Wq    = (const float*)d_in[1];
  const float* Wk    = (const float*)d_in[2];
  const float* Wv    = (const float*)d_in[3];
  const float* Wa    = (const float*)d_in[4];
  const float* Wb    = (const float*)d_in[5];
  const float* Wg    = (const float*)d_in[6];
  const float* Wo    = (const float*)d_in[7];
  const float* cwq   = (const float*)d_in[8];
  const float* cwk   = (const float*)d_in[9];
  const float* cwv   = (const float*)d_in[10];
  const float* decay = (const float*)d_in[11];
  const float* A_log = (const float*)d_in[12];
  const float* Dp    = (const float*)d_in[13];
  const float* dtb   = (const float*)d_in[14];
  const float* onw   = (const float*)d_in[15];

  char* wsb = (char*)d_ws;
  size_t off = 0;
  auto alloc = [&](size_t bytes) {
    char* p = wsb + off;
    off += (bytes + 255) & ~(size_t)255;
    return p;
  };
  unsigned short* hsb = (unsigned short*)alloc((size_t)BT * DM * 2);
  unsigned* wqp = (unsigned*)alloc((size_t)KD * (DM / 2) * 4);
  unsigned* wkp = (unsigned*)alloc((size_t)KD * (DM / 2) * 4);
  unsigned* wvp = (unsigned*)alloc((size_t)VD * (DM / 2) * 4);
  unsigned* wgp = (unsigned*)alloc((size_t)VD * (DM / 2) * 4);
  unsigned* wop = (unsigned*)alloc((size_t)DM * (VD / 2) * 4);
  float* q_pre   = (float*)alloc((size_t)BT * KD * 4);
  float* k_pre   = (float*)alloc((size_t)BT * KD * 4);
  float* v_pre   = (float*)alloc((size_t)BT * VD * 4);
  float* gatebuf = (float*)alloc((size_t)BT * VD * 4);
  float* aw      = (float*)alloc((size_t)BT * Hh * 4);
  float* bw      = (float*)alloc((size_t)BT * Hh * 4);
  float* q_conv  = (float*)alloc((size_t)BT * KD * 4);
  float* k_conv  = (float*)alloc((size_t)BT * KD * 4);
  float* v_conv  = (float*)alloc((size_t)BT * VD * 4);
  unsigned short* qnb = (unsigned short*)alloc((size_t)BT * KD * 2);
  unsigned short* knb = (unsigned short*)alloc((size_t)BT * KD * 2);
  unsigned short* pnb = (unsigned short*)alloc((size_t)BT * KD * 2);
  unsigned* ppkT = (unsigned*)alloc((size_t)KD * (BT / 2) * 4);
  unsigned* vpkT = (unsigned*)alloc((size_t)VD * (BT / 2) * 4);
  float* gv    = (float*)alloc((size_t)BT * Hh * 4);
  float* bev   = (float*)alloc((size_t)BT * Hh * 4);
  float* gCbuf = (float*)alloc((size_t)Bc * Hh * NC * 4);
  unsigned short* Mbuf = (unsigned short*)alloc((size_t)Bc * Hh * NC * CH * CH * 2);
  unsigned short* W2b  = (unsigned short*)alloc((size_t)Bc * Hh * NC * CH * HK * 2);
  unsigned short* Qg   = (unsigned short*)alloc((size_t)Bc * Hh * NC * CH * HK * 2);
  unsigned short* KpT  = (unsigned short*)alloc((size_t)Bc * Hh * NC * HK * CH * 2);
  float* Ubase = (float*)alloc((size_t)Bc * Hh * NC * CH * HV * 4);
  float* obuf  = (float*)alloc((size_t)BT * VD * 4);
  unsigned short* out2b = (unsigned short*)alloc((size_t)BT * VD * 2);

  dim3 blk(256);
  auto g1d = [](long n) { return dim3((unsigned)((n + 255) / 256)); };

  // one-time bf16 conversions / packing (B operands transposed pair-packed)
  k_f2bf<<<g1d((long)BT * DM), blk, 0, stream>>>(hs, hsb, (long)BT * DM);
  k_packwT<<<g1d((long)KD * (DM / 2)), blk, 0, stream>>>(Wq, wqp, DM, KD);
  k_packwT<<<g1d((long)KD * (DM / 2)), blk, 0, stream>>>(Wk, wkp, DM, KD);
  k_packwT<<<g1d((long)VD * (DM / 2)), blk, 0, stream>>>(Wv, wvp, DM, VD);
  k_packwT<<<g1d((long)VD * (DM / 2)), blk, 0, stream>>>(Wg, wgp, DM, VD);
  k_packwT<<<g1d((long)DM * (VD / 2)), blk, 0, stream>>>(Wo, wop, VD, DM);
  // projections (bf16 WMMA)
  k_gemm_bf<KD, DM><<<dim3(KD / 128, BT / 64), blk, 0, stream>>>(hsb, wqp, q_pre);
  k_gemm_bf<KD, DM><<<dim3(KD / 128, BT / 64), blk, 0, stream>>>(hsb, wkp, k_pre);
  k_gemm_bf<VD, DM><<<dim3(VD / 128, BT / 64), blk, 0, stream>>>(hsb, wvp, v_pre);
  k_gemm_bf<VD, DM><<<dim3(VD / 128, BT / 64), blk, 0, stream>>>(hsb, wgp, gatebuf);
  k_gemm_small<<<g1d((long)BT * Hh), blk, 0, stream>>>(hs, Wa, aw, BT, Hh, DM);
  k_gemm_small<<<g1d((long)BT * Hh), blk, 0, stream>>>(hs, Wb, bw, BT, Hh, DM);
  // short convs + SiLU
  k_conv_silu<<<g1d((long)BT * KD), blk, 0, stream>>>(q_pre, cwq, q_conv, KD);
  k_conv_silu<<<g1d((long)BT * KD), blk, 0, stream>>>(k_pre, cwk, k_conv, KD);
  k_conv_silu<<<g1d((long)BT * VD), blk, 0, stream>>>(v_pre, cwv, v_conv, VD);
  // per-head features (bf16 outputs)
  k_features<<<BT * Hh, blk, 0, stream>>>(q_conv, k_conv, decay, Dp, A_log, dtb,
                                          aw, bw, qnb, knb, pnb, gv, bev);
  // transposed time-pair packing for K4 B-operands
  k_packrowsT<<<g1d((long)KD * (BT / 2)), blk, 0, stream>>>(pnb, ppkT, BT, KD);
  k_packwT<<<g1d((long)VD * (BT / 2)), blk, 0, stream>>>(v_conv, vpkT, BT, VD);
  // parallel chunk precompute
  size_t lds4 = (size_t)(384 + 2 * CH * CH) * sizeof(float);  // 34,304 B
  k_chunkprep<<<Bc * Hh * NC, blk, lds4, stream>>>(qnb, knb, pnb, ppkT, vpkT, gv,
                                                   bev, gCbuf, Mbuf, W2b, Qg,
                                                   KpT, Ubase);
  // sequential recurrence: 96-way parallel; S fp32 + bf16 shadow + TDM stage
  size_t lds5 = 65536 + 32768 + 8192 + 2 * (size_t)CH * HK * 2;  // 172,032 B
  k_recur<<<Bc * Hh * 8, blk, lds5, stream>>>(Mbuf, W2b, Qg, KpT, Ubase, gCbuf,
                                              obuf);
  // gated RMSNorm (bf16 out) + output projection
  k_gatednorm<<<BT * Hh, blk, 0, stream>>>(obuf, gatebuf, onw, out2b);
  k_gemm_bf<DM, VD><<<dim3(DM / 128, BT / 64), blk, 0, stream>>>(out2b, wop,
                                                                 (float*)d_out);
}
